// MoeSparseExpertsLayer_13331578487343
// MI455X (gfx1250) — matrix operand order repaired and checked by reading, not compiled, not run on hardware
//
#include <hip/hip_runtime.h>

typedef __attribute__((ext_vector_type(16))) __bf16 v16bf;
typedef __attribute__((ext_vector_type(8)))  float  v8f;

__device__ inline v8f zero8() {
  v8f z;
#pragma unroll
  for (int i = 0; i < 8; ++i) z[i] = 0.0f;
  return z;
}

// ---------------------------------------------------------------------------
// Small utility kernels
// ---------------------------------------------------------------------------
__global__ void zero_counts_kernel(int* counts) {
  if (threadIdx.x < 8) counts[threadIdx.x] = 0;
}

__global__ void scan_kernel(const int* __restrict__ counts, int* __restrict__ offsets) {
  if (threadIdx.x == 0 && blockIdx.x == 0) {
    int acc = 0;
    for (int e = 0; e < 8; ++e) { offsets[e] = acc; acc += counts[e]; }
    offsets[8] = acc;
  }
}

__global__ __launch_bounds__(256) void build_lists_kernel(
    const int* __restrict__ topk_idx, const int* __restrict__ topk_rank,
    const float* __restrict__ topk_w, const int* __restrict__ offsets,
    int* __restrict__ expert_tok, float* __restrict__ expert_wt, int total) {
  int id = blockIdx.x * 256 + threadIdx.x;
  if (id >= total) return;
  int e    = topk_idx[id];
  int slot = offsets[e] + topk_rank[id];
  expert_tok[slot] = id >> 1;      // token id
  expert_wt[slot]  = topk_w[id];
}

// Convert fp32 [mats][K][N] weights to bf16 in WMMA B-fragment layout:
// dst[frag*512 + lane*16 + i] = W[ks*32 + (lane>>4)*8 + (i&7) + (i>>3)*16][nt*16 + (lane&15)]
// with frag = ks*(N/16) + nt.
__global__ __launch_bounds__(256) void swizzle_kernel(
    const float* __restrict__ src, __bf16* __restrict__ dst, int K, int N, int mats) {
  long long total = (long long)mats * K * N;
  long long id = (long long)blockIdx.x * 256 + threadIdx.x;
  if (id >= total) return;
  int matKN = K * N;
  int mat = (int)(id / matKN);
  int r   = (int)(id - (long long)mat * matKN);
  int i    = r & 15;
  int lane = (r >> 4) & 31;
  int frag = r >> 9;
  int ntilesN = N >> 4;
  int nt = frag % ntilesN;
  int ks = frag / ntilesN;
  int k = ks * 32 + ((lane >> 4) << 3) + (i & 7) + ((i >> 3) << 4);
  int n = nt * 16 + (lane & 15);
  dst[id] = (__bf16)src[(size_t)mat * matKN + (size_t)k * N + n];
}

// ---------------------------------------------------------------------------
// Router: logits, softmax, top-2, shared-gate sigmoid. One wave per token.
// ---------------------------------------------------------------------------
__global__ __launch_bounds__(256) void router_kernel(
    const float* __restrict__ x, const float* __restrict__ gate_w,
    const float* __restrict__ sgate, float* __restrict__ logits_out,
    int* __restrict__ topk_idx, int* __restrict__ topk_rank,
    float* __restrict__ topk_w, float* __restrict__ sig,
    int* __restrict__ counts, int T) {
  int lane = threadIdx.x & 31;
  int t = blockIdx.x * 8 + (threadIdx.x >> 5);
  if (t >= T) return;

  float p[8];
  float ps = 0.0f;
#pragma unroll
  for (int e = 0; e < 8; ++e) p[e] = 0.0f;

  for (int h = lane; h < 512; h += 32) {
    float xv = x[(size_t)t * 512 + h];
#pragma unroll
    for (int e = 0; e < 8; ++e) p[e] += xv * gate_w[h * 8 + e];
    ps += xv * sgate[h];
  }
#pragma unroll
  for (int e = 0; e < 8; ++e) {
#pragma unroll
    for (int off = 16; off > 0; off >>= 1) p[e] += __shfl_xor(p[e], off, 32);
  }
#pragma unroll
  for (int off = 16; off > 0; off >>= 1) ps += __shfl_xor(ps, off, 32);

  if (lane == 0) {
    float mx = p[0];
#pragma unroll
    for (int e = 1; e < 8; ++e) mx = fmaxf(mx, p[e]);
    float ex[8];
    float s = 0.0f;
#pragma unroll
    for (int e = 0; e < 8; ++e) { ex[e] = __expf(p[e] - mx); s += ex[e]; }
    float inv = 1.0f / s;
    int i1 = 0; float b1 = -1.0f;
#pragma unroll
    for (int e = 0; e < 8; ++e) { float pr = ex[e] * inv; if (pr > b1) { b1 = pr; i1 = e; } }
    int i2 = 0; float b2 = -1.0f;
#pragma unroll
    for (int e = 0; e < 8; ++e) {
      if (e == i1) continue;
      float pr = ex[e] * inv;
      if (pr > b2) { b2 = pr; i2 = e; }
    }
#pragma unroll
    for (int e = 0; e < 8; ++e) logits_out[(size_t)t * 8 + e] = p[e];
    topk_idx[2 * t]     = i1;
    topk_idx[2 * t + 1] = i2;
    topk_w[2 * t]       = b1;
    topk_w[2 * t + 1]   = b2;
    sig[t] = 1.0f / (1.0f + __expf(-ps));
    topk_rank[2 * t]     = atomicAdd(&counts[i1], 1);
    topk_rank[2 * t + 1] = atomicAdd(&counts[i2], 1);
  }
}

// ---------------------------------------------------------------------------
// Fused SwiGLU FFN tile kernel. 32-token tiles, 8 waves (256 threads).
// Each wave: stage 1 computes two 16-row M-tiles against one 16-col inter
// tile (4 independent WMMA chains, B fragments reused across 32 tokens);
// stage 2 owns 64 H-columns x 2 M-tiles (8 persistent v8f accumulators).
// mode 0: gathered expert tokens -> weighted bf16 rows in moe_buf[slot]
// mode 1: sequential tokens, shared expert -> out = sig * ffn (fp32)
// ---------------------------------------------------------------------------
__global__ __launch_bounds__(256) void ffn_kernel(
    const float* __restrict__ x,
    const __bf16* __restrict__ w1b, const __bf16* __restrict__ w3b,
    const __bf16* __restrict__ w2b,
    int inter, int mode,
    const int* __restrict__ expert_tok, const float* __restrict__ expert_wt,
    const int* __restrict__ offsets, const int* __restrict__ counts,
    const float* __restrict__ sig,
    __bf16* __restrict__ moe_buf, float* __restrict__ out, int T) {
  __shared__ __align__(16) __bf16 xtile[32][520];   // 512 cols + pad (16B-aligned rows)
  __shared__ __align__(16) __bf16 htile[32][136];   // 128 cols + pad
  __shared__ int   toks[32];
  __shared__ float wrow[32];

  int tid = threadIdx.x;
  int valid = 32, s0 = 0, t0 = 0;
  const __bf16 *w1e, *w3e, *w2e;

  if (mode == 0) {
    int e = blockIdx.y;
    int cnt = counts[e];
    int tile0 = blockIdx.x * 32;
    if (tile0 >= cnt) return;                       // uniform exit, no barrier yet
    int base = offsets[e];
    if (tid < 32) {
      int r  = tile0 + tid;
      int rc = (r < cnt) ? r : (cnt - 1);
      toks[tid] = expert_tok[base + rc];
      wrow[tid] = (r < cnt) ? expert_wt[base + r] : 0.0f;
    }
    valid = cnt - tile0; if (valid > 32) valid = 32;
    s0 = base + tile0;
    w1e = w1b + (size_t)e * 512 * 1024;
    w3e = w3b + (size_t)e * 512 * 1024;
    w2e = w2b + (size_t)e * 1024 * 512;
  } else {
    t0 = blockIdx.x * 32;
    if (tid < 32) { toks[tid] = t0 + tid; wrow[tid] = sig[t0 + tid]; }
    w1e = w1b; w3e = w3b; w2e = w2b;
  }
  __syncthreads();

  // Stage 32 x-rows (fp32 -> bf16) into LDS, coalesced float4 loads.
  for (int it = 0; it < 16; ++it) {
    int eidx = it * 256 + tid;          // 4096 float4 total
    int row = eidx >> 7;
    int c4  = eidx & 127;
    float4 f = ((const float4*)(x + (size_t)toks[row] * 512))[c4];
    int cc = c4 * 4;
    xtile[row][cc]     = (__bf16)f.x;
    xtile[row][cc + 1] = (__bf16)f.y;
    xtile[row][cc + 2] = (__bf16)f.z;
    xtile[row][cc + 3] = (__bf16)f.w;
  }
  __syncthreads();

  int lane = tid & 31;
  int wv   = tid >> 5;                  // 8 waves
  int mcol = lane & 15;
  int ksel = lane >> 4;
  int itiles  = inter >> 4;
  int nchunks = inter >> 7;             // 128-wide inter chunks

  v8f accO[8];                          // [nt*2 + mtile]
#pragma unroll
  for (int q = 0; q < 8; ++q) accO[q] = zero8();

  for (int c = 0; c < nchunks; ++c) {
    // ---- stage 1: a1/a3 for this wave's 16 inter-cols, 2 M-tiles ----
    v8f a10 = zero8(), a11 = zero8(), a30 = zero8(), a31 = zero8();
    int ntg = c * 8 + wv;
    for (int ks = 0; ks < 16; ++ks) {
      v16bf a0, a1, b1, b3;
      ((uint4*)&a0)[0] = *(const uint4*)&xtile[mcol][ks * 32 + 8 * ksel];
      ((uint4*)&a0)[1] = *(const uint4*)&xtile[mcol][ks * 32 + 16 + 8 * ksel];
      ((uint4*)&a1)[0] = *(const uint4*)&xtile[mcol + 16][ks * 32 + 8 * ksel];
      ((uint4*)&a1)[1] = *(const uint4*)&xtile[mcol + 16][ks * 32 + 16 + 8 * ksel];
      size_t fb = (((size_t)(ks * itiles + ntg)) << 9) + lane * 16;
      ((uint4*)&b1)[0] = *(const uint4*)(w1e + fb);
      ((uint4*)&b1)[1] = *(const uint4*)(w1e + fb + 8);
      ((uint4*)&b3)[0] = *(const uint4*)(w3e + fb);
      ((uint4*)&b3)[1] = *(const uint4*)(w3e + fb + 8);
      a10 = __builtin_amdgcn_wmma_f32_16x16x32_bf16(false, a0, false, b1,
                                                    (short)0, a10, false, false);
      a11 = __builtin_amdgcn_wmma_f32_16x16x32_bf16(false, a1, false, b1,
                                                    (short)0, a11, false, false);
      a30 = __builtin_amdgcn_wmma_f32_16x16x32_bf16(false, a0, false, b3,
                                                    (short)0, a30, false, false);
      a31 = __builtin_amdgcn_wmma_f32_16x16x32_bf16(false, a1, false, b3,
                                                    (short)0, a31, false, false);
    }
    // silu(a1)*a3 -> bf16 h-tile in LDS (both M-tiles)
#pragma unroll
    for (int j = 0; j < 8; ++j) {
      int row = j + 8 * ksel;
      float v1 = a10[j], v3 = a30[j];
      float hv = v1 / (1.0f + __expf(-v1)) * v3;
      htile[row][wv * 16 + mcol] = (__bf16)hv;
      float w1v = a11[j], w3v = a31[j];
      float hw = w1v / (1.0f + __expf(-w1v)) * w3v;
      htile[row + 16][wv * 16 + mcol] = (__bf16)hw;
    }
    __syncthreads();
    // ---- stage 2: out += h @ W2 (each wave owns 64 H-columns, 2 M-tiles) ----
#pragma unroll
    for (int ks2 = 0; ks2 < 4; ++ks2) {
      v16bf ha0, ha1;
      ((uint4*)&ha0)[0] = *(const uint4*)&htile[mcol][ks2 * 32 + 8 * ksel];
      ((uint4*)&ha0)[1] = *(const uint4*)&htile[mcol][ks2 * 32 + 16 + 8 * ksel];
      ((uint4*)&ha1)[0] = *(const uint4*)&htile[mcol + 16][ks2 * 32 + 8 * ksel];
      ((uint4*)&ha1)[1] = *(const uint4*)&htile[mcol + 16][ks2 * 32 + 16 + 8 * ksel];
      int ksg = c * 4 + ks2;
#pragma unroll
      for (int nt = 0; nt < 4; ++nt) {
        v16bf b2;
        size_t fb2 = (((size_t)(ksg * 32 + wv * 4 + nt)) << 9) + lane * 16;
        ((uint4*)&b2)[0] = *(const uint4*)(w2e + fb2);
        ((uint4*)&b2)[1] = *(const uint4*)(w2e + fb2 + 8);
        accO[nt * 2]     = __builtin_amdgcn_wmma_f32_16x16x32_bf16(
            false, ha0, false, b2, (short)0, accO[nt * 2], false, false);
        accO[nt * 2 + 1] = __builtin_amdgcn_wmma_f32_16x16x32_bf16(
            false, ha1, false, b2, (short)0, accO[nt * 2 + 1], false, false);
      }
    }
    __syncthreads();
  }

  // Epilogue: apply per-row weight, write rows.
#pragma unroll
  for (int nt = 0; nt < 4; ++nt) {
#pragma unroll
    for (int mt = 0; mt < 2; ++mt) {
#pragma unroll
      for (int j = 0; j < 8; ++j) {
        int row = mt * 16 + j + 8 * ksel;
        int col = wv * 64 + nt * 16 + mcol;
        float v = accO[nt * 2 + mt][j] * wrow[row];
        if (mode == 0) {
          if (row < valid)
            moe_buf[(((size_t)(s0 + row)) << 9) + col] = (__bf16)v;
        } else {
          out[(((size_t)(t0 + row)) << 9) + col] = v;
        }
      }
    }
  }
}

// ---------------------------------------------------------------------------
// Combine: out[t][h] += moe_buf[slot0][h] + moe_buf[slot1][h]
// ---------------------------------------------------------------------------
__global__ __launch_bounds__(256) void combine_kernel(
    float* __restrict__ out, const __bf16* __restrict__ moe_buf,
    const int* __restrict__ topk_idx, const int* __restrict__ topk_rank,
    const int* __restrict__ offsets, long long total) {
  long long id = (long long)blockIdx.x * 256 + threadIdx.x;
  if (id >= total) return;
  int t = (int)(id >> 9);
  int h = (int)(id & 511);
  int i0 = 2 * t, i1 = 2 * t + 1;
  int s0 = offsets[topk_idx[i0]] + topk_rank[i0];
  int s1 = offsets[topk_idx[i1]] + topk_rank[i1];
  out[id] += (float)moe_buf[(((size_t)s0) << 9) + h] +
             (float)moe_buf[(((size_t)s1) << 9) + h];
}

// ---------------------------------------------------------------------------
extern "C" void kernel_launch(void* const* d_in, const int* in_sizes, int n_in,
                              void* d_out, int out_size, void* d_ws, size_t ws_size,
                              hipStream_t stream) {
  const float* x      = (const float*)d_in[0];
  const float* gate_w = (const float*)d_in[1];
  const float* w1     = (const float*)d_in[2];
  const float* w2     = (const float*)d_in[3];
  const float* w3     = (const float*)d_in[4];
  const float* sw1    = (const float*)d_in[5];
  const float* sw2    = (const float*)d_in[6];
  const float* sw3    = (const float*)d_in[7];
  const float* sgate  = (const float*)d_in[8];

  const int H = 512, E = 8, MI = 1024, SI = 2048;
  const int T = in_sizes[0] / H;   // 49152

  float* out        = (float*)d_out;
  float* logits_out = out + (size_t)T * H;

  // Workspace carve (~134 MB)
  char* p = (char*)d_ws;
  auto carve = [&](size_t bytes) -> void* {
    void* q = (void*)p;
    p += (bytes + 255) & ~(size_t)255;
    return q;
  };
  __bf16* w1b  = (__bf16*)carve((size_t)E * H * MI * 2);
  __bf16* w3b  = (__bf16*)carve((size_t)E * H * MI * 2);
  __bf16* w2b  = (__bf16*)carve((size_t)E * MI * H * 2);
  __bf16* sw1b = (__bf16*)carve((size_t)H * SI * 2);
  __bf16* sw3b = (__bf16*)carve((size_t)H * SI * 2);
  __bf16* sw2b = (__bf16*)carve((size_t)SI * H * 2);
  int*   topk_idx   = (int*)carve((size_t)T * 2 * 4);
  int*   topk_rank  = (int*)carve((size_t)T * 2 * 4);
  float* topk_w     = (float*)carve((size_t)T * 2 * 4);
  float* sig        = (float*)carve((size_t)T * 4);
  int*   counts     = (int*)carve(64);
  int*   offsets    = (int*)carve(64);
  int*   expert_tok = (int*)carve((size_t)T * 2 * 4);
  float* expert_wt  = (float*)carve((size_t)T * 2 * 4);
  __bf16* moe_buf   = (__bf16*)carve((size_t)T * 2 * H * 2);

  zero_counts_kernel<<<1, 32, 0, stream>>>(counts);

  auto launch_sw = [&](const float* s, __bf16* d, int K, int N, int mats) {
    long long tot = (long long)mats * K * N;
    swizzle_kernel<<<(int)((tot + 255) / 256), 256, 0, stream>>>(s, d, K, N, mats);
  };
  launch_sw(w1,  w1b,  H,  MI, E);
  launch_sw(w3,  w3b,  H,  MI, E);
  launch_sw(w2,  w2b,  MI, H,  E);
  launch_sw(sw1, sw1b, H,  SI, 1);
  launch_sw(sw3, sw3b, H,  SI, 1);
  launch_sw(sw2, sw2b, SI, H,  1);

  router_kernel<<<T / 8, 256, 0, stream>>>(x, gate_w, sgate, logits_out,
                                           topk_idx, topk_rank, topk_w, sig, counts, T);
  scan_kernel<<<1, 32, 0, stream>>>(counts, offsets);
  build_lists_kernel<<<(2 * T + 255) / 256, 256, 0, stream>>>(
      topk_idx, topk_rank, topk_w, offsets, expert_tok, expert_wt, 2 * T);

  // Shared expert: out = sigmoid(x@sgate) * ffn_shared(x)
  ffn_kernel<<<dim3(T / 32, 1), 256, 0, stream>>>(
      x, sw1b, sw3b, sw2b, SI, 1,
      nullptr, nullptr, nullptr, nullptr, sig, nullptr, out, T);

  // Routed experts: weighted rows into moe_buf (slot-indexed; deterministic)
  ffn_kernel<<<dim3((T + 31) / 32, E), 256, 0, stream>>>(
      x, w1b, w3b, w2b, MI, 0,
      expert_tok, expert_wt, offsets, counts, nullptr, moe_buf, out, T);

  combine_kernel<<<(int)(((long long)T * H + 255) / 256), 256, 0, stream>>>(
      out, moe_buf, topk_idx, topk_rank, offsets, (long long)T * H);
}
